// MNISTBinarizedModel_49074296324140
// MI455X (gfx1250) — compile-verified, matrix operand order checked by reading, not run on hardware
//
#include <hip/hip_runtime.h>

typedef __attribute__((ext_vector_type(8))) int v8i;

#define BATCH 32768
#define EPSBN 1e-5f

// ---------------------------------------------------------------------------
// Binarize input x (threshold 0.5, since layer0 input is x-0.5) into packed
// int8 {+1,-1}, K zero-padded.  One block per row.
// ---------------------------------------------------------------------------
__global__ void pack_x_kernel(const float* __restrict__ x, char* __restrict__ dst,
                              int kin, int kpad_dw) {
  int row = blockIdx.x;
  int c = threadIdx.x;
  if (c >= kpad_dw) return;
  const float* xr = x + (size_t)row * kin;
  unsigned int w = 0;
#pragma unroll
  for (int j = 0; j < 4; ++j) {
    int kk = c * 4 + j;
    int byte = 0;
    if (kk < kin) byte = (xr[kk] >= 0.5f) ? 1 : 0xFF;  // +1 / -1
    w |= (unsigned)(byte & 0xFF) << (8 * j);
  }
  ((unsigned int*)dst)[(size_t)row * kpad_dw + c] = w;
}

// Binarize weights (threshold 0) into packed int8 {+1,-1}; rows >= out_rows
// and k >= kin padded with 0 (contributes nothing to the integer dot product).
__global__ void pack_w_kernel(const float* __restrict__ W, char* __restrict__ dst,
                              int out_rows, int kin, int kpad_dw) {
  int row = blockIdx.x;
  int c = threadIdx.x;
  if (c >= kpad_dw) return;
  unsigned int w = 0;
  if (row < out_rows) {
    const float* wr = W + (size_t)row * kin;
#pragma unroll
    for (int j = 0; j < 4; ++j) {
      int kk = c * 4 + j;
      int byte = 0;
      if (kk < kin) byte = (wr[kk] >= 0.0f) ? 1 : 0xFF;
      w |= (unsigned)(byte & 0xFF) << (8 * j);
    }
  }
  ((unsigned int*)dst)[(size_t)row * kpad_dw + c] = w;
}

__global__ void zero_stats_kernel(unsigned long long* s, unsigned long long* q) {
  s[threadIdx.x] = 0ull;
  q[threadIdx.x] = 0ull;
}

// ---------------------------------------------------------------------------
// dword offset inside a 16-row x 64-byte tile for 8-bit WMMA fragments
// (ISA 7.12.2, 8-bit A 16x64 layout; same K striping used for B columns).
// ---------------------------------------------------------------------------
__device__ __forceinline__ int frag_kdw(int v, int half) {
  return ((v >= 4) ? 8 : 0) + (((v & 3) >> 1) << 2) + (v & 1) + half * 2;
}

// 80-byte LDS row stride: 16B-aligned (allows ds_store_b128 commits) and
// bank-conflict-free for 16-row fragment reads (gcd(20,64)=4, cycle 16).
#define LDA_STRIDE 20

// ---------------------------------------------------------------------------
// H[64 x 256 tile] = A(+-1 int8) @ W(+-1 int8)^T via v_wmma_i32_16x16x64_iu8.
// 512 threads = 16 waves: wm in {0,1} x wn in {0..7}; wave tile 32x32
// (2x2 16x16 tiles, 32 acc VGPRs).  Staged-load addressing uses uniform
// base (saddr) + one 32-bit byte offset per stream, bumped by 64B per kstep,
// to minimize address VGPRs so the staged tiles stay in registers.
// Double-buffered LDS, one barrier per kstep; staged global loads for ks+1
// are in flight during the WMMAs of ks.  Per-column sum/sumsq folded
// lane<->lane+16, int32 LDS atomics, one global u64 atomic per column.
// ---------------------------------------------------------------------------
__global__ void __launch_bounds__(512) gemm_i8_n256(
    const char* __restrict__ Apk, const char* __restrict__ Wpk,
    int* __restrict__ H, unsigned long long* __restrict__ colsum,
    unsigned long long* __restrict__ colssq, int kpad_dw) {
  __shared__ int lA[2 * 64 * LDA_STRIDE];
  __shared__ int lB[2 * 256 * LDA_STRIDE];
  __shared__ int lsum[256], lssq[256];

  const int tid = threadIdx.x;
  const int lane = tid & 31;
  const int wave = tid >> 5;   // 0..15
  const int wm = wave >> 3;    // 0..1  (M)
  const int wn = wave & 7;     // 0..7  (N)
  const int half = lane >> 4;
  const int l16 = lane & 15;
  const int rowBase = blockIdx.x * 64;

  if (tid < 256) { lsum[tid] = 0; lssq[tid] = 0; }

  v8i acc[2][2];
#pragma unroll
  for (int mt = 0; mt < 2; ++mt)
#pragma unroll
    for (int nt = 0; nt < 2; ++nt)
#pragma unroll
      for (int j = 0; j < 8; ++j) acc[mt][nt][j] = 0;

  const int ksteps = kpad_dw >> 4;  // 64 bytes of K per step

  // staging split across 512 threads:
  // A tile 64x16 dw -> b64 per thread; B tile 256x16 dw -> 2x b128 per thread
  const int ar = tid >> 3;        // 0..63
  const int ac = (tid & 7) * 2;   // dword offset, 8B aligned
  const int br = tid >> 2;        // 0..127
  const int bc = (tid & 3) * 4;   // dword offset, 16B aligned

  // 32-bit byte offsets from the uniform bases (saddr+voffset addressing)
  unsigned aoff = (unsigned)((rowBase + ar) * kpad_dw + ac) * 4u;
  unsigned boff0 = (unsigned)(br * kpad_dw + bc) * 4u;
  unsigned boff1 = (unsigned)((br + 128) * kpad_dw + bc) * 4u;

  uint2 aS;
  uint4 bS0, bS1;
  // preload kstep 0 into registers
  aS = *(const uint2*)(Apk + aoff);
  bS0 = *(const uint4*)(Wpk + boff0);
  bS1 = *(const uint4*)(Wpk + boff1);

  for (int ks = 0; ks < ksteps; ++ks) {
    const int buf = ks & 1;
    int* la = lA + buf * 64 * LDA_STRIDE;
    int* lb = lB + buf * 256 * LDA_STRIDE;

    // commit staged registers to LDS
    *(uint2*)&la[ar * LDA_STRIDE + ac] = aS;
    *(uint4*)&lb[br * LDA_STRIDE + bc] = bS0;
    *(uint4*)&lb[(br + 128) * LDA_STRIDE + bc] = bS1;
    __syncthreads();

    // stage kstep ks+1 (overlaps with the WMMAs below)
    if (ks + 1 < ksteps) {
      aoff += 64u;
      boff0 += 64u;
      boff1 += 64u;
      aS = *(const uint2*)(Apk + aoff);
      bS0 = *(const uint4*)(Wpk + boff0);
      bS1 = *(const uint4*)(Wpk + boff1);
      if (ks + 2 < ksteps)  // global_prefetch_b8: pull streaming A tile early
        __builtin_prefetch(Apk + aoff + 64u, 0, 1);
    }

    // fragments + WMMA from current buffer
    v8i afrag[2];
#pragma unroll
    for (int mt = 0; mt < 2; ++mt) {
      int row = wm * 32 + mt * 16 + l16;
#pragma unroll
      for (int v = 0; v < 8; ++v)
        afrag[mt][v] = la[row * LDA_STRIDE + frag_kdw(v, half)];
    }
#pragma unroll
    for (int nt = 0; nt < 2; ++nt) {
      int col = wn * 32 + nt * 16 + l16;
      v8i bfrag;
#pragma unroll
      for (int v = 0; v < 8; ++v)
        bfrag[v] = lb[col * LDA_STRIDE + frag_kdw(v, half)];
#pragma unroll
      for (int mt = 0; mt < 2; ++mt) {
        acc[mt][nt] = __builtin_amdgcn_wmma_i32_16x16x64_iu8(
            true, afrag[mt], true, bfrag, acc[mt][nt], false, false);
      }
    }
  }

  // epilogue: store H (int32) and reduce per-column sum / sumsq (int32 exact:
  // per-column per-WG bounds: |sum| <= 64*784, ssq <= 64*784^2 < 2^31)
#pragma unroll
  for (int mt = 0; mt < 2; ++mt) {
#pragma unroll
    for (int nt = 0; nt < 2; ++nt) {
      int col = wn * 32 + nt * 16 + l16;
      int s = 0, q = 0;
#pragma unroll
      for (int j = 0; j < 8; ++j) {
        int h = acc[mt][nt][j];
        int row = rowBase + wm * 32 + mt * 16 + half * 8 + j;  // C/D layout
        H[(size_t)row * 256 + col] = h;
        s += h;
        q += h * h;
      }
      // lanes L and L+16 hold the same column (rows j and j+8): fold them
      s += __shfl_xor(s, 16);
      q += __shfl_xor(q, 16);
      if (half == 0) {
        atomicAdd(&lsum[col], s);
        atomicAdd(&lssq[col], q);
      }
    }
  }
  __syncthreads();
  if (tid < 256) {
    atomicAdd(&colsum[tid], (unsigned long long)(long long)lsum[tid]);
    atomicAdd(&colssq[tid], (unsigned long long)(long long)lssq[tid]);
  }
}

// ---------------------------------------------------------------------------
// Final layer: N padded to 16.  8 waves each own 32 rows x 16 cols; WG = 256
// rows.  Same double-buffered pipeline with 32-bit offset addressing.
// ---------------------------------------------------------------------------
__global__ void __launch_bounds__(256) gemm_i8_n16(
    const char* __restrict__ Apk, const char* __restrict__ Wpk,
    int* __restrict__ H, unsigned long long* __restrict__ colsum,
    unsigned long long* __restrict__ colssq, int kpad_dw) {
  __shared__ int lA[2 * 256 * LDA_STRIDE];
  __shared__ int lB[2 * 16 * LDA_STRIDE];
  __shared__ int lsum[16], lssq[16];

  const int tid = threadIdx.x;
  const int lane = tid & 31;
  const int wave = tid >> 5;
  const int half = lane >> 4;
  const int l16 = lane & 15;
  const int rowBase = blockIdx.x * 256;

  if (tid < 16) { lsum[tid] = 0; lssq[tid] = 0; }

  v8i acc[2];
#pragma unroll
  for (int mt = 0; mt < 2; ++mt)
#pragma unroll
    for (int j = 0; j < 8; ++j) acc[mt][j] = 0;

  const int ksteps = kpad_dw >> 4;

  const int sr = tid >> 2;
  const int sc = (tid & 3) * 4;

  unsigned aoff[4];
#pragma unroll
  for (int i = 0; i < 4; ++i)
    aoff[i] = (unsigned)((rowBase + sr + i * 64) * kpad_dw + sc) * 4u;
  unsigned boff = (unsigned)(sr * kpad_dw + sc) * 4u;

  uint4 aS[4], bS;
#pragma unroll
  for (int i = 0; i < 4; ++i) aS[i] = *(const uint4*)(Apk + aoff[i]);
  if (tid < 64) bS = *(const uint4*)(Wpk + boff);

  for (int ks = 0; ks < ksteps; ++ks) {
    const int buf = ks & 1;
    int* la = lA + buf * 256 * LDA_STRIDE;
    int* lb = lB + buf * 16 * LDA_STRIDE;

#pragma unroll
    for (int i = 0; i < 4; ++i)
      *(uint4*)&la[(sr + i * 64) * LDA_STRIDE + sc] = aS[i];
    if (tid < 64) *(uint4*)&lb[sr * LDA_STRIDE + sc] = bS;
    __syncthreads();

    if (ks + 1 < ksteps) {
#pragma unroll
      for (int i = 0; i < 4; ++i) {
        aoff[i] += 64u;
        aS[i] = *(const uint4*)(Apk + aoff[i]);
      }
      boff += 64u;
      if (tid < 64) bS = *(const uint4*)(Wpk + boff);
    }

    v8i bfrag;
#pragma unroll
    for (int v = 0; v < 8; ++v)
      bfrag[v] = lb[l16 * LDA_STRIDE + frag_kdw(v, half)];
#pragma unroll
    for (int mt = 0; mt < 2; ++mt) {
      int row = wave * 32 + mt * 16 + l16;
      v8i afrag;
#pragma unroll
      for (int v = 0; v < 8; ++v)
        afrag[v] = la[row * LDA_STRIDE + frag_kdw(v, half)];
      acc[mt] = __builtin_amdgcn_wmma_i32_16x16x64_iu8(
          true, afrag, true, bfrag, acc[mt], false, false);
    }
  }

#pragma unroll
  for (int mt = 0; mt < 2; ++mt) {
    int s = 0, q = 0;
#pragma unroll
    for (int j = 0; j < 8; ++j) {
      int h = acc[mt][j];
      int row = rowBase + wave * 32 + mt * 16 + half * 8 + j;
      H[(size_t)row * 16 + l16] = h;
      s += h;
      q += h * h;
    }
    s += __shfl_xor(s, 16);
    q += __shfl_xor(q, 16);
    if (half == 0) {
      atomicAdd(&lsum[l16], s);
      atomicAdd(&lssq[l16], q);
    }
  }
  __syncthreads();
  if (tid < 16) {
    atomicAdd(&colsum[tid], (unsigned long long)(long long)lsum[tid]);
    atomicAdd(&colssq[tid], (unsigned long long)(long long)lssq[tid]);
  }
}

// ---------------------------------------------------------------------------
// BN stats -> per-column affine: bn(h) = h*scale + shift (exact int64 sums).
// ---------------------------------------------------------------------------
__global__ void bn_stats_kernel(const unsigned long long* __restrict__ cs,
                                const unsigned long long* __restrict__ cq,
                                const float* __restrict__ g, const float* __restrict__ b,
                                float* __restrict__ ss, int ncols, int ngb) {
  int c = threadIdx.x;
  if (c >= ncols) return;
  long long s = (long long)cs[c];
  long long q = (long long)cq[c];
  double mean = (double)s / (double)BATCH;
  double var = (double)q / (double)BATCH - mean * mean;
  float gg = (c < ngb) ? g[c] : 1.0f;
  float bb = (c < ngb) ? b[c] : 0.0f;
  float inv = rsqrtf((float)var + EPSBN);
  float scale = gg * inv;
  ss[c] = scale;
  ss[256 + c] = bb - (float)mean * scale;
}

// BN + binarize fused: next activation byte = (h*scale+shift >= 0) ? +1 : -1
__global__ void bn_pack_kernel(const int* __restrict__ H, const float* __restrict__ ss,
                               char* __restrict__ dst) {
  __shared__ float s_scale[256], s_shift[256];
  int t = threadIdx.x;
  s_scale[t] = ss[t];
  s_shift[t] = ss[256 + t];
  __syncthreads();
  int idx = blockIdx.x * 256 + t;          // one dword (4 cols) per thread
  int4 h4 = ((const int4*)H)[idx];
  int c0 = (idx & 63) * 4;
  unsigned int w = 0;
  int hv[4] = {h4.x, h4.y, h4.z, h4.w};
#pragma unroll
  for (int j = 0; j < 4; ++j) {
    float v = (float)hv[j] * s_scale[c0 + j] + s_shift[c0 + j];
    int byte = (v >= 0.0f) ? 1 : 0xFF;
    w |= (unsigned)(byte & 0xFF) << (8 * j);
  }
  ((unsigned int*)dst)[idx] = w;
}

// BN + softmax over the 10 real output columns (H stored as B x 16).
__global__ void softmax_kernel(const int* __restrict__ H4, const float* __restrict__ ss,
                               float* __restrict__ out) {
  int row = blockIdx.x * blockDim.x + threadIdx.x;
  float y[10];
  float m = -3.4e38f;
#pragma unroll
  for (int j = 0; j < 10; ++j) {
    y[j] = (float)H4[(size_t)row * 16 + j] * ss[j] + ss[256 + j];
    m = fmaxf(m, y[j]);
  }
  float sum = 0.0f;
#pragma unroll
  for (int j = 0; j < 10; ++j) {
    y[j] = __expf(y[j] - m);
    sum += y[j];
  }
  float inv = 1.0f / sum;
#pragma unroll
  for (int j = 0; j < 10; ++j) out[(size_t)row * 10 + j] = y[j] * inv;
}

// ---------------------------------------------------------------------------
extern "C" void kernel_launch(void* const* d_in, const int* in_sizes, int n_in,
                              void* d_out, int out_size, void* d_ws, size_t ws_size,
                              hipStream_t stream) {
  (void)in_sizes; (void)n_in; (void)out_size; (void)ws_size;
  const float* x = (const float*)d_in[0];
  const float* W[5]; const float* g[5]; const float* b[5];
  for (int i = 0; i < 5; ++i) {
    W[i] = (const float*)d_in[1 + 3 * i];
    g[i] = (const float*)d_in[2 + 3 * i];
    b[i] = (const float*)d_in[3 + 3 * i];
  }

  char* ws = (char*)d_ws;
  size_t off = 0;
  auto take = [&](size_t bytes) -> char* {
    char* p = ws + off;
    off = (off + bytes + 255) & ~(size_t)255;
    return p;
  };
  const size_t B = BATCH;
  char* A0 = take(B * 832);        // layer-0 activations, K padded 784->832
  char* A1 = take(B * 256);        // ping
  char* A2 = take(B * 256);        // pong
  char* Wp[5];
  Wp[0] = take((size_t)256 * 832);
  Wp[1] = take((size_t)256 * 256);
  Wp[2] = take((size_t)256 * 256);
  Wp[3] = take((size_t)256 * 256);
  Wp[4] = take((size_t)16 * 256);  // 10 rows padded to 16
  int* H = (int*)take(B * 256 * 4);
  unsigned long long* colsum = (unsigned long long*)take(256 * 8);
  unsigned long long* colssq = (unsigned long long*)take(256 * 8);
  float* ss = (float*)take(512 * 4);

  // binarize input + all weights
  pack_x_kernel<<<BATCH, 256, 0, stream>>>(x, A0, 784, 208);
  pack_w_kernel<<<256, 256, 0, stream>>>(W[0], Wp[0], 256, 784, 208);
  pack_w_kernel<<<256, 256, 0, stream>>>(W[1], Wp[1], 256, 256, 64);
  pack_w_kernel<<<256, 256, 0, stream>>>(W[2], Wp[2], 256, 256, 64);
  pack_w_kernel<<<256, 256, 0, stream>>>(W[3], Wp[3], 256, 256, 64);
  pack_w_kernel<<<16, 256, 0, stream>>>(W[4], Wp[4], 10, 256, 64);

  const char* Acur = A0;
  char* Anext = A1;
  int kdw = 208;
  for (int k = 0; k < 4; ++k) {
    zero_stats_kernel<<<1, 256, 0, stream>>>(colsum, colssq);
    gemm_i8_n256<<<BATCH / 64, 512, 0, stream>>>(Acur, Wp[k], H, colsum, colssq, kdw);
    bn_stats_kernel<<<1, 256, 0, stream>>>(colsum, colssq, g[k], b[k], ss, 256, 256);
    bn_pack_kernel<<<BATCH * 64 / 256, 256, 0, stream>>>(H, ss, Anext);
    Acur = Anext;
    Anext = (Anext == A1) ? A2 : A1;
    kdw = 64;
  }
  zero_stats_kernel<<<1, 256, 0, stream>>>(colsum, colssq);
  gemm_i8_n16<<<BATCH / 256, 256, 0, stream>>>(Acur, Wp[4], H, colsum, colssq, 64);
  bn_stats_kernel<<<1, 256, 0, stream>>>(colsum, colssq, g[4], b[4], ss, 16, 10);
  softmax_kernel<<<BATCH / 256, 256, 0, stream>>>(H, ss, (float*)d_out);
}
